// AttForwardTA_47631187313265
// MI455X (gfx1250) — matrix operand order repaired and checked by reading, compiled, not run on hardware
//
#include <hip/hip_runtime.h>
#include <hip/hip_bf16.h>
#include <math.h>

// ---------------- types ----------------
typedef __attribute__((ext_vector_type(16))) __bf16 v16bf;
typedef __attribute__((ext_vector_type(8)))  __bf16 v8bf;
typedef __attribute__((ext_vector_type(8)))  float  v8f;

#define B_   64
#define T_   1000
#define EU_  512
#define DU_  1024
#define AD_  128
#define AC_  32
#define KW_  31
#define OD_  80
#define TAIN_ (EU_ + DU_ + OD_)   // 1616
#define NTILE_T 63                // ceil(1000/16)
#define ASTRIDE 520               // padded bf16 row stride for A in LDS (16B aligned, bank-staggered)

// round-to-nearest-even float -> bf16 bits
__device__ __forceinline__ unsigned short f2bf(float f) {
    union { float f; unsigned u; } x; x.f = f;
    unsigned u = x.u;
    u += 0x7FFFu + ((u >> 16) & 1u);
    return (unsigned short)(u >> 16);
}

// ---------------- kernel 1: prep (w_enc->bf16 transposed, F = conv_w . w_att, dec_t = dec_z@w_dec + b_enc)
__global__ __launch_bounds__(256) void k_prep(const float* __restrict__ w_enc,
                                              const float* __restrict__ b_enc,
                                              const float* __restrict__ w_dec,
                                              const float* __restrict__ w_att,
                                              const float* __restrict__ conv_w,
                                              const float* __restrict__ dec_z,
                                              unsigned short* __restrict__ w_encT,
                                              float* __restrict__ Fbuf,
                                              float* __restrict__ dec_t) {
    int g = blockIdx.x * 256 + threadIdx.x;
    if (g < EU_ * AD_) {
        // w_encT[n][k] = bf16(w_enc[k][n]) ; n-major rows of length 512
        int n = g >> 9, k = g & 511;
        w_encT[g] = f2bf(w_enc[k * AD_ + n]);
    } else if (g < EU_ * AD_ + KW_ * AD_) {
        int i = g - EU_ * AD_;
        int k = i / AD_, n = i % AD_;
        float acc = 0.f;
        #pragma unroll
        for (int c = 0; c < AC_; ++c)
            acc = fmaf(conv_w[c * KW_ + k], w_att[c * AD_ + n], acc);
        Fbuf[i] = acc;
    } else if (g < EU_ * AD_ + KW_ * AD_ + B_ * AD_) {
        int i = g - (EU_ * AD_ + KW_ * AD_);
        int b = i >> 7, n = i & 127;
        float acc = b_enc[n];
        for (int k = 0; k < DU_; ++k)
            acc = fmaf(dec_z[b * DU_ + k], w_dec[k * AD_ + n], acc);
        dec_t[i] = acc;
    }
}

// ---------------- kernel 2: fused pre(WMMA-bf16) + loc-conv + dec_t + tanh + w_g dot -> e[b,t]
__global__ __launch_bounds__(256) void k_e(const float* __restrict__ enc,
                                           const unsigned short* __restrict__ w_encT,
                                           const float* __restrict__ Fbuf,
                                           const float* __restrict__ dec_t,
                                           const float* __restrict__ att_prev,
                                           const float* __restrict__ w_g,
                                           const float* __restrict__ b_g,
                                           float* __restrict__ e_out) {
    __shared__ unsigned short Abf[16 * ASTRIDE]; // 16 rows x 512 bf16 (padded)
    __shared__ float Fl[KW_ * AD_];              // 31x128 combined filter
    __shared__ float sl[48];                     // att_prev slice [t0-15 .. t0+30]
    __shared__ float dn[AD_];                    // dec_t row (+b_enc folded)
    __shared__ float etile[16];

    const int tid = threadIdx.x;
    const int b  = blockIdx.y;
    const int t0 = blockIdx.x * 16;

    // stage A tile: enc[b, t0..t0+15, 0..511] -> bf16 LDS (coalesced)
    for (int i = tid; i < 16 * EU_; i += 256) {
        int r = i >> 9, c = i & 511;
        int t = t0 + r;
        float v = (t < T_) ? enc[((long)b * T_ + t) * EU_ + c] : 0.f;
        Abf[r * ASTRIDE + c] = f2bf(v);
    }
    for (int i = tid; i < KW_ * AD_; i += 256) Fl[i] = Fbuf[i];
    if (tid < AD_) dn[tid] = dec_t[b * AD_ + tid];
    if (tid < 46) {
        int t = t0 - 15 + tid;
        sl[tid] = (t >= 0 && t < T_) ? att_prev[b * T_ + t] : 0.f;
    }
    if (tid < 48 && tid >= 46) sl[tid] = 0.f;
    if (tid < 16) etile[tid] = 0.f;
    __syncthreads();

    const int wave = tid >> 5;          // 0..7 -> N tile
    const int lane = tid & 31;
    const int half = lane >> 4;         // 0: lanes 0-15, 1: lanes 16-31
    const int lrow = lane & 15;
    const int n0   = wave * 16;
    const int n    = n0 + lrow;

    // A per-lane (16-bit 16x32 layout): row M=lrow; K = half*8 + {0..7}, then +16
    const unsigned short* arow = &Abf[lrow * ASTRIDE + half * 8];
    // B per-lane (16-bit 32x16 layout): column N=n; K = half*8 + {0..7}, then +16 (same as A)
    const unsigned short* brow = &w_encT[(long)n * EU_ + half * 8];

    v8f acc = {};
    #pragma unroll 4
    for (int ks = 0; ks < 16; ++ks) {
        const int kb = ks * 32;
        v8bf a0 = *(const v8bf*)(arow + kb);
        v8bf a1 = *(const v8bf*)(arow + kb + 16);
        v16bf A = __builtin_shufflevector(a0, a1, 0,1,2,3,4,5,6,7,8,9,10,11,12,13,14,15);
        v8bf b0 = *(const v8bf*)(brow + kb);
        v8bf b1 = *(const v8bf*)(brow + kb + 16);
        v16bf Bv = __builtin_shufflevector(b0, b1, 0,1,2,3,4,5,6,7,8,9,10,11,12,13,14,15);
        acc = __builtin_amdgcn_wmma_f32_16x16x32_bf16(false, A, false, Bv,
                                                      (short)0, acc, false, false);
    }

    // epilogue: + dec_t + location conv, tanh, w_g-weighted reduce over n
    const float wg = w_g[n];
    const int Mbase = half * 8;
    #pragma unroll
    for (int r = 0; r < 8; ++r) {
        const int M = Mbase + r;
        const int t = t0 + M;
        float v = acc[r] + dn[n];
        float ac = 0.f;
        #pragma unroll
        for (int k = 0; k < KW_; ++k)
            ac = fmaf(Fl[k * AD_ + n], sl[M + k], ac);
        v += ac;
        float contrib = (t < T_) ? wg * tanhf(v) : 0.f;
        // reduce across the 16 lanes of this half (xor masks < 16 stay in-half)
        #pragma unroll
        for (int m = 1; m < 16; m <<= 1)
            contrib += __shfl_xor(contrib, m, 32);
        if (lrow == 0) atomicAdd(&etile[M], contrib);
    }
    __syncthreads();
    if (tid < 16) {
        int t = t0 + tid;
        if (t < T_) e_out[b * T_ + t] = etile[tid] + b_g[0];
    }
}

// ---------------- kernel 3: masked softmax + trans-agent blend + clamp + L1 normalize
__global__ __launch_bounds__(256) void k_w(const float* __restrict__ e_buf,
                                           const int* __restrict__ lens,
                                           const float* __restrict__ att_prev,
                                           float* __restrict__ out_w,
                                           float* __restrict__ out_c) {
    __shared__ float wbuf[T_];
    __shared__ float red[256];
    const int b = blockIdx.x, tid = threadIdx.x;
    const int len = lens[b];

    // max
    float mx = -INFINITY;
    for (int t = tid; t < T_; t += 256) {
        float v = (t < len) ? e_buf[b * T_ + t] : -INFINITY;
        mx = fmaxf(mx, v);
    }
    red[tid] = mx; __syncthreads();
    for (int s = 128; s > 0; s >>= 1) { if (tid < s) red[tid] = fmaxf(red[tid], red[tid + s]); __syncthreads(); }
    const float gmax = red[0]; __syncthreads();

    // exp + sum
    float sm = 0.f;
    for (int t = tid; t < T_; t += 256) {
        float ev = (t < len) ? expf(e_buf[b * T_ + t] - gmax) : 0.f;
        wbuf[t] = ev; sm += ev;
    }
    red[tid] = sm; __syncthreads();
    for (int s = 128; s > 0; s >>= 1) { if (tid < s) red[tid] += red[tid + s]; __syncthreads(); }
    const float S = red[0]; __syncthreads();
    const float invS = 1.f / S;

    // blend with 0.5*att_prev + 0.5*shift, clamp(1e-6) everywhere, sum
    float s2 = 0.f;
    for (int t = tid; t < T_; t += 256) {
        float w0  = wbuf[t] * invS;
        float ap  = att_prev[b * T_ + t];
        float aps = (t > 0) ? att_prev[b * T_ + t - 1] : 0.f;
        float wv  = fmaxf(0.5f * (ap + aps) * w0, 1e-6f);
        wbuf[t] = wv; s2 += wv;
    }
    red[tid] = s2; __syncthreads();
    for (int s = 128; s > 0; s >>= 1) { if (tid < s) red[tid] += red[tid + s]; __syncthreads(); }
    const float inv2 = 1.f / fmaxf(red[0], 1e-12f);
    __syncthreads();

    for (int t = tid; t < T_; t += 256) out_w[b * T_ + t] = wbuf[t] * inv2;
    // zero the context output region for k_c's atomics
    for (int i = tid; i < EU_; i += 256) out_c[b * EU_ + i] = 0.f;
}

// ---------------- kernel 4: context c = sum_t w[t]*enc[b,t,:] (8 T-chunks, atomic combine)
__global__ __launch_bounds__(256) void k_c(const float* __restrict__ enc,
                                           const float* __restrict__ w,
                                           float* __restrict__ out_c) {
    const int b = blockIdx.y;
    const int ts = blockIdx.x * 125;
    const int te = ts + 125;
    const int c0 = threadIdx.x, c1 = threadIdx.x + 256;
    float a0 = 0.f, a1 = 0.f;
    for (int t = ts; t < te; ++t) {
        const float wt = w[b * T_ + t];
        const float* row = enc + ((long)b * T_ + t) * EU_;
        __builtin_prefetch(row + EU_, 0, 1);
        a0 = fmaf(wt, row[c0], a0);
        a1 = fmaf(wt, row[c1], a1);
    }
    atomicAdd(&out_c[b * EU_ + c0], a0);
    atomicAdd(&out_c[b * EU_ + c1], a1);
}

// ---------------- kernel 5: transition agent prob = sigmoid([c,out_prev,dec_z] @ w_ta + b_ta)
__global__ __launch_bounds__(256) void k_ta(const float* __restrict__ c,
                                            const float* __restrict__ out_prev,
                                            const float* __restrict__ dec_z,
                                            const float* __restrict__ w_ta,
                                            const float* __restrict__ b_ta,
                                            float* __restrict__ prob) {
    __shared__ float red[256];
    const int b = blockIdx.x, tid = threadIdx.x;
    float p = 0.f;
    for (int i = tid; i < TAIN_; i += 256) {
        float x;
        if (i < EU_)            x = c[b * EU_ + i];
        else if (i < EU_ + OD_) x = out_prev[b * OD_ + (i - EU_)];
        else                    x = dec_z[b * DU_ + (i - EU_ - OD_)];
        p = fmaf(x, w_ta[i], p);
    }
    red[tid] = p; __syncthreads();
    for (int s = 128; s > 0; s >>= 1) { if (tid < s) red[tid] += red[tid + s]; __syncthreads(); }
    if (tid == 0) prob[b] = 1.f / (1.f + expf(-(red[0] + b_ta[0])));
}

// ---------------- launcher ----------------
extern "C" void kernel_launch(void* const* d_in, const int* in_sizes, int n_in,
                              void* d_out, int out_size, void* d_ws, size_t ws_size,
                              hipStream_t stream) {
    const float* enc      = (const float*)d_in[0];   // (64,1000,512)
    const int*   lens     = (const int*)  d_in[1];   // (64)
    const float* dec_z    = (const float*)d_in[2];   // (64,1024)
    const float* att_prev = (const float*)d_in[3];   // (64,1000)
    const float* out_prev = (const float*)d_in[4];   // (64,80)
    const float* w_enc    = (const float*)d_in[5];   // (512,128)
    const float* b_enc    = (const float*)d_in[6];   // (128)
    const float* w_dec    = (const float*)d_in[7];   // (1024,128)
    const float* w_att    = (const float*)d_in[8];   // (32,128)
    const float* conv_w   = (const float*)d_in[9];   // (32,1,31)
    const float* w_g      = (const float*)d_in[10];  // (128)
    const float* b_g      = (const float*)d_in[11];  // scalar
    const float* w_ta     = (const float*)d_in[12];  // (1616,1)
    const float* b_ta     = (const float*)d_in[13];  // (1)

    // outputs: c (64*512) | w (64*1000) | prob (64)
    float* out   = (float*)d_out;
    float* out_c = out;
    float* out_w = out + B_ * EU_;
    float* out_p = out + B_ * EU_ + B_ * T_;

    // workspace layout (~436 KB)
    char* ws = (char*)d_ws;
    unsigned short* w_encT = (unsigned short*)ws;            // 512*128*2   = 131072 B
    float* Fbuf  = (float*)(ws + 131072);                    // 31*128*4    =  15872 B
    float* dec_t = (float*)(ws + 147456);                    // 64*128*4    =  32768 B
    float* e_buf = (float*)(ws + 180224);                    // 64*1000*4   = 256000 B

    // 1) prep: 65536 + 3968 + 8192 = 77696 work items
    k_prep<<<dim3(304), dim3(256), 0, stream>>>(w_enc, b_enc, w_dec, w_att, conv_w,
                                                dec_z, w_encT, Fbuf, dec_t);
    // 2) fused WMMA energy kernel: (63 T-tiles) x (64 batches)
    k_e<<<dim3(NTILE_T, B_), dim3(256), 0, stream>>>(enc, w_encT, Fbuf, dec_t,
                                                     att_prev, w_g, b_g, e_buf);
    // 3) softmax/blend/normalize per batch
    k_w<<<dim3(B_), dim3(256), 0, stream>>>(e_buf, lens, att_prev, out_w, out_c);
    // 4) context accumulation: 8 chunks x 64 batches
    k_c<<<dim3(8, B_), dim3(256), 0, stream>>>(enc, out_w, out_c);
    // 5) transition agent
    k_ta<<<dim3(B_), dim3(256), 0, stream>>>(out_c, out_prev, dec_z, w_ta, b_ta, out_p);
}
